// ResidualAttentionBlock_48112223650326
// MI455X (gfx1250) — matrix operand order repaired and checked by reading, not compiled, not run on hardware
//
#include <hip/hip_runtime.h>
#include <math.h>
#include <stdint.h>

typedef __attribute__((ext_vector_type(16))) _Float16 v16h;
typedef __attribute__((ext_vector_type(8)))  float    v8f;

#define T 512
#define D 128
#define H 8
#define B 8
#define QT 16          // query tile rows
#define VCHUNK 64      // keys per staged V panel
#define HALF_SUN 4.5f
#define INV_SQRT_D 0.08838834764831845f  // 1/sqrt(128)

// ---------------------------------------------------------------------------
// WMMA fragment index helpers (CDNA5 16x16x32 f16 layouts, wave32)
// A (16x32, MxK): lane L -> M = L&15 ; elem i -> K = (i<8?0:16) + (L>>4)*8 + (i&7)
// B (32x16, KxN): lane L -> N = L&15 ; elem i -> K = (i<8?0:16) + (L>>4)*8 + (i&7)
// C/D (16x16 f32): vgpr r -> M = r + 8*(L>>4) ; N = L&15
// ---------------------------------------------------------------------------
__device__ __forceinline__ int a_kidx(int i, int hk) {
    return ((i < 8) ? 0 : 16) + hk * 8 + (i & 7);
}
__device__ __forceinline__ int b_kidx(int i, int hk) {
    return ((i < 8) ? 0 : 16) + hk * 8 + (i & 7);
}

// ===========================================================================
// Kernel 1: QV projection  y[bt][o] = sum_i wqv[o][i]*x[bt][i] + wqv[o][128]
//   x: (4096,128) f32, wqv: (2048,129), y: (4096,2048) f32
//   grid (256,16), block 256 (8 waves); wave -> one 16x16 tile, 8 N-tiles/WG
// ===========================================================================
__global__ void __launch_bounds__(256)
qv_gemm_kernel(const float* __restrict__ x, const float* __restrict__ wqv,
               float* __restrict__ y) {
    const int lane = threadIdx.x & 31;
    const int wave = threadIdx.x >> 5;
    const int m_base = blockIdx.x * 16;
    const int n_base = (blockIdx.y * 8 + wave) * 16;
    const int M  = lane & 15;
    const int hk = lane >> 4;

    v8f acc = {};
#pragma unroll
    for (int k0 = 0; k0 < D; k0 += 32) {
        v16h a, b;
#pragma unroll
        for (int i = 0; i < 16; ++i) {
            a[i] = (_Float16)x[(m_base + M) * D + k0 + a_kidx(i, hk)];
            b[i] = (_Float16)wqv[(n_base + M) * (D + 1) + k0 + b_kidx(i, hk)];
        }
        acc = __builtin_amdgcn_wmma_f32_16x16x32_f16(false, a, false, b,
                                                     (short)0, acc, false, false);
    }
#pragma unroll
    for (int r = 0; r < 8; ++r) {
        const int row = m_base + r + 8 * hk;
        const int o   = n_base + (lane & 15);
        y[row * 2048 + o] = acc[r] + wqv[o * (D + 1) + D];
    }
}

// ===========================================================================
// Kernel 2: fused L1 attention + softmax*mask + PV(WMMA) + quick_gelu shift
//   One WG per (b, head, 16-row query tile).  2048 WGs, 256 threads (8 waves).
//   q tile staged via global_load_async_to_lds_b128 (ASYNCcnt path);
//   V panel staged to LDS as f16 in 64-key chunks; WMMA fed entirely from LDS.
// ===========================================================================
__global__ void __launch_bounds__(256)
attn_kernel(const float* __restrict__ x, const float* __restrict__ msk,
            const float* __restrict__ wk, const float* __restrict__ y,
            float* __restrict__ yo) {
    __shared__ __align__(16) float    qs[QT][D];        //  8 KB  q tile
    __shared__ __align__(16) float    wks[D];           // 0.5KB per-head k weights
    __shared__ __align__(16) float    sc[QT][T];        // 32 KB scores / exp
    __shared__ __align__(16) _Float16 pf[QT][T];        // 16 KB probabilities (f16)
    __shared__ __align__(16) _Float16 vf[VCHUNK][D];    // 16 KB staged V panel (f16)

    const int tid = threadIdx.x;
    const int wg  = blockIdx.x;
    const int qt  = wg & 31;           // query tile
    const int h_  = (wg >> 5) & 7;     // head
    const int b_  = wg >> 8;           // batch

    // ---- stage q tile (16 rows x 512B) via async copy to LDS ---------------
#pragma unroll
    for (int it = 0; it < 2; ++it) {
        const int c  = it * 256 + tid;          // 512 16-byte chunks
        const int r  = c >> 5;
        const int c4 = (c & 31) * 4;
        const float* gsrc =
            &y[((size_t)(b_ * T + qt * QT + r) * H + h_) * 256 + c4];
        const unsigned lds_off = (unsigned)(uintptr_t)&qs[r][c4];
        const unsigned long long gaddr = (unsigned long long)(uintptr_t)gsrc;
        asm volatile("global_load_async_to_lds_b128 %0, %1, off"
                     :: "v"(lds_off), "v"(gaddr) : "memory");
    }
    if (tid < D) wks[tid] = wk[h_ * D + tid];
    asm volatile("s_wait_asynccnt 0x0" ::: "memory");
    __syncthreads();

    // ---- L1 scores: thread handles keys tid and tid+256 --------------------
#pragma unroll
    for (int kk = 0; kk < 2; ++kk) {
        const int s = kk * 256 + tid;
        const float* xrow = x + (size_t)(b_ * T + s) * D;
        __builtin_prefetch(xrow, 0, 0);              // global_prefetch_b8
        float acc[QT];
#pragma unroll
        for (int r = 0; r < QT; ++r) acc[r] = 0.f;
        for (int c0 = 0; c0 < D; c0 += 4) {
            const float4 xv = *(const float4*)(xrow + c0);
            const float k0 = xv.x * wks[c0 + 0];
            const float k1 = xv.y * wks[c0 + 1];
            const float k2 = xv.z * wks[c0 + 2];
            const float k3 = xv.w * wks[c0 + 3];
#pragma unroll
            for (int r = 0; r < QT; ++r) {
                const float4 qv = *(const float4*)(&qs[r][c0]);
                acc[r] += fabsf(qv.x - k0) + fabsf(qv.y - k1) +
                          fabsf(qv.z - k2) + fabsf(qv.w - k3);
            }
        }
#pragma unroll
        for (int r = 0; r < QT; ++r) sc[r][s] = -acc[r] * INV_SQRT_D;
    }
    __syncthreads();

    // ---- softmax over 512 keys: 16 lanes per row (wave32, width-16 shfl) ---
    {
        const int r = tid >> 4, sub = tid & 15;
        float m = -3.4e38f;
        for (int c = sub; c < T; c += 16) m = fmaxf(m, sc[r][c]);
#pragma unroll
        for (int off = 8; off >= 1; off >>= 1) m = fmaxf(m, __shfl_xor(m, off, 16));
        float sum = 0.f;
        for (int c = sub; c < T; c += 16) {
            const float e = __expf(sc[r][c] - m);
            sc[r][c] = e;
            sum += e;
        }
#pragma unroll
        for (int off = 8; off >= 1; off >>= 1) sum += __shfl_xor(sum, off, 16);
        const float inv = 1.f / sum;
        const int tq = qt * QT + r;
        const float* mrow = msk + ((size_t)h_ * T + tq) * T;
        for (int c = sub; c < T; c += 16)
            pf[r][c] = (_Float16)(sc[r][c] * inv * mrow[c]);
    }

    // ---- PV: (16x512)x(512x16) per wave, WMMA fed from LDS -----------------
    const int lane = tid & 31;
    const int wave = tid >> 5;
    const int d0   = wave * 16;            // 8 waves cover d = 0..127
    const int M    = lane & 15;
    const int hk   = lane >> 4;

    v8f acc = {};
    for (int s0 = 0; s0 < T; s0 += VCHUNK) {
        __syncthreads();
        // cooperatively stage V panel (64 keys x 128 cols) as f16, coalesced
#pragma unroll
        for (int e = tid; e < VCHUNK * D; e += 256) {
            const int sr = e >> 7, c = e & 127;
            vf[sr][c] = (_Float16)
                y[((size_t)(b_ * T + s0 + sr) * H + h_) * 256 + D + c];
        }
        __syncthreads();
#pragma unroll
        for (int k0 = 0; k0 < VCHUNK; k0 += 32) {
            v16h a, b;
#pragma unroll
            for (int i = 0; i < 16; ++i) {
                a[i] = pf[M][s0 + k0 + a_kidx(i, hk)];
                b[i] = vf[k0 + b_kidx(i, hk)][d0 + M];
            }
            acc = __builtin_amdgcn_wmma_f32_16x16x32_f16(false, a, false, b,
                                                         (short)0, acc, false, false);
        }
    }
    // quick_gelu(bo + 4.5) - 4.5, write yo workspace
#pragma unroll
    for (int r = 0; r < 8; ++r) {
        const int tq = qt * QT + r + 8 * hk;
        const float g = acc[r] + HALF_SUN;
        const float v = g / (1.f + __expf(-1.702f * g)) - HALF_SUN;
        yo[(size_t)(b_ * T + tq) * D + d0 + (lane & 15)] = v;
    }
}

// ===========================================================================
// Kernel 3: fanout + residual  out = x + yo @ fwᵀ + bias
//   (4096x128)x(128x128); grid 256, block 256 (8 waves = 8 N-tiles)
// ===========================================================================
__global__ void __launch_bounds__(256)
fanout_kernel(const float* __restrict__ x, const float* __restrict__ fw,
              const float* __restrict__ yo, float* __restrict__ out) {
    const int lane = threadIdx.x & 31;
    const int wave = threadIdx.x >> 5;
    const int m_base = blockIdx.x * 16;
    const int n_base = wave * 16;
    const int M  = lane & 15;
    const int hk = lane >> 4;

    v8f acc = {};
#pragma unroll
    for (int k0 = 0; k0 < D; k0 += 32) {
        v16h a, b;
#pragma unroll
        for (int i = 0; i < 16; ++i) {
            a[i] = (_Float16)yo[(m_base + M) * D + k0 + a_kidx(i, hk)];
            b[i] = (_Float16)fw[(n_base + M) * (D + 1) + k0 + b_kidx(i, hk)];
        }
        acc = __builtin_amdgcn_wmma_f32_16x16x32_f16(false, a, false, b,
                                                     (short)0, acc, false, false);
    }
#pragma unroll
    for (int r = 0; r < 8; ++r) {
        const int row = m_base + r + 8 * hk;
        const int o   = n_base + (lane & 15);
        out[row * D + o] = x[row * D + o] + acc[r] + fw[o * (D + 1) + D];
    }
}

// ===========================================================================
extern "C" void kernel_launch(void* const* d_in, const int* in_sizes, int n_in,
                              void* d_out, int out_size, void* d_ws, size_t ws_size,
                              hipStream_t stream) {
    const float* x    = (const float*)d_in[0];  // (8,512,128)
    const float* msk  = (const float*)d_in[1];  // (1,8,512,512)
    const float* wqv  = (const float*)d_in[2];  // (2048,129)
    const float* wk   = (const float*)d_in[3];  // (8,128)
    const float* fw   = (const float*)d_in[4];  // (128,129)
    float* out = (float*)d_out;                 // (8,512,128)

    float* y_ws  = (float*)d_ws;                        // 4096*2048 f32 = 32 MB
    float* yo_ws = y_ws + (size_t)B * T * H * 2 * D;    // 4096*128  f32 =  2 MB

    // 1) QV projection GEMM (WMMA f16->f32)
    qv_gemm_kernel<<<dim3(256, 16), 256, 0, stream>>>(x, wqv, y_ws);
    // 2) fused L1-attention + softmax*mask + PV (WMMA) + shifted quick-gelu
    attn_kernel<<<B * H * (T / QT), 256, 0, stream>>>(x, msk, wk, y_ws, yo_ws);
    // 3) fanout GEMM + bias + residual (WMMA)
    fanout_kernel<<<T * B / 16, 256, 0, stream>>>(x, fw, yo_ws, out);
}
